// SGC_55834574848373
// MI455X (gfx1250) — matrix operand order repaired and checked by reading, not compiled, not run on hardware
//
#include <hip/hip_runtime.h>
#include <hip/hip_bf16.h>
#include <stdint.h>

typedef __attribute__((ext_vector_type(16))) __bf16 v16bf;
typedef __attribute__((ext_vector_type(8)))  __bf16 v8bf;
typedef __attribute__((ext_vector_type(8)))  float  v8f;
typedef __attribute__((ext_vector_type(4)))  int    v4i;

#if defined(__HIP_DEVICE_COMPILE__) && \
    __has_builtin(__builtin_amdgcn_global_load_async_to_lds_b128) && \
    __has_builtin(__builtin_amdgcn_s_wait_asynccnt)
#define USE_ASYNC_LDS 1
typedef __attribute__((address_space(1))) v4i gv4i;   // global int4
typedef __attribute__((address_space(3))) v4i lv4i;   // LDS int4
#else
#define USE_ASYNC_LDS 0
#endif

// ---------------- degree / normalization ----------------
__global__ void k_fill1(float* __restrict__ d, int n) {
  int i = blockIdx.x * blockDim.x + threadIdx.x;
  if (i < n) d[i] = 1.0f;   // self-loop contributes 1 to every node's degree
}

__global__ void k_deg(const long long* __restrict__ dst, float* __restrict__ deg, int E) {
  int e = blockIdx.x * blockDim.x + threadIdx.x;
  if (e < E) atomicAdd(&deg[(int)dst[e]], 1.0f);
}

__global__ void k_rsqrt(float* __restrict__ d, int n) {
  int i = blockIdx.x * blockDim.x + threadIdx.x;
  if (i < n) d[i] = rsqrtf(d[i]);   // deg >= 1 always (self-loops)
}

// ---------------- data staging ----------------
// emb (N x 56, rows 224B -> float4 aligned) into A with stride 64, zero pad
__global__ void k_copy_emb(const float* __restrict__ emb, float* __restrict__ A, int n) {
  int i = blockIdx.x * blockDim.x + threadIdx.x;   // over n*16 float4 slots
  if (i >= n * 16) return;
  int node = i >> 4, c = i & 15;
  float4 v = make_float4(0.f, 0.f, 0.f, 0.f);
  if (c < 14) v = ((const float4*)emb)[node * 14 + c];
  ((float4*)A)[i] = v;
}

// W (rows x kin, f32) -> bf16 (rows x kout), zero pad k >= kin
__global__ void k_cvt_w(const float* __restrict__ w, __bf16* __restrict__ wb,
                        int rows, int kin, int kout) {
  int i = blockIdx.x * blockDim.x + threadIdx.x;
  if (i >= rows * kout) return;
  int r = i / kout, k = i - r * kout;
  wb[i] = (k < kin) ? (__bf16)w[r * kin + k] : (__bf16)0.0f;
}

// ---------------- propagation (one hop) ----------------
// self-loop term + full overwrite of destination buffer (incl. zero pads)
__global__ void k_prop_init(const float* __restrict__ x, float* __restrict__ y,
                            const float* __restrict__ dinv, int n, int S4) {
  int i = blockIdx.x * blockDim.x + threadIdx.x;   // over n * S/4 float4 slots
  if (i >= n * S4) return;
  int node = i / S4;
  float w = dinv[node]; w *= w;
  float4 v = ((const float4*)x)[i];
  float4 r; r.x = w * v.x; r.y = w * v.y; r.z = w * v.z; r.w = w * v.w;
  ((float4*)y)[i] = r;
}

// edge-parallel scatter: thread = (edge, 4-feature chunk); L2-resident atomics
__global__ void k_prop_edges(const long long* __restrict__ src,
                             const long long* __restrict__ dst,
                             const float* __restrict__ dinv,
                             const float* __restrict__ x, float* __restrict__ y,
                             int E, int C, int S) {
  int i = blockIdx.x * blockDim.x + threadIdx.x;
  if (i >= E * C) return;
  int e = i / C, c = i - e * C;
  int s = (int)src[e], d = (int)dst[e];
  float w = dinv[s] * dinv[d];
  float4 v = *(const float4*)(x + (long long)s * S + c * 4);
  float* yp = y + (long long)d * S + c * 4;
  atomicAdd(yp + 0, w * v.x);
  atomicAdd(yp + 1, w * v.y);
  atomicAdd(yp + 2, w * v.z);
  atomicAdd(yp + 3, w * v.w);
}

// ---------------- WMMA GEMM: Y = X @ Wb^T + b ----------------
// One wave -> one 16x16 output tile; block = 8 waves -> 128 output cols.
// X tile (16 rows x K f32) staged once per block in LDS (async path if available),
// weights pre-converted to bf16. KT = K/32 (compile-time) -> fully unrolled WMMAs.
template<int KT>
__global__ void __launch_bounds__(256)
k_gemm_bf16(const float* __restrict__ X, int ldx,
            const __bf16* __restrict__ Wb,
            const float* __restrict__ bias,
            float* __restrict__ Y, int ldy, int n) {
  constexpr int K = 32 * KT;
  __shared__ float tile[16 * K];

  const int tid  = threadIdx.x;
  const int lane = tid & 31;
  const int wave = tid >> 5;
  const int m    = lane & 15;    // A row / B col / D col
  const int h    = lane >> 4;    // lane-half selector
  const int row0 = blockIdx.x * 16;
  const int col0 = wave << 4;

  // ---- cooperative stage of the X tile into LDS ----
  {
    constexpr int nchunks = 16 * K / 4;     // float4 chunks (256 or 512)
    #pragma unroll
    for (int c0 = 0; c0 < nchunks; c0 += 256) {
      int c = c0 + tid;
      int r = c / (K / 4), q = c - r * (K / 4);
      int rr = row0 + r; if (rr >= n) rr = n - 1;   // clamp (n%16==0 in practice)
      const float* gp = X + (size_t)rr * ldx + q * 4;
      float* lp = &tile[r * K + q * 4];
#if USE_ASYNC_LDS
      __builtin_amdgcn_global_load_async_to_lds_b128((gv4i*)gp, (lv4i*)lp, 0, 0);
#else
      *(float4*)lp = *(const float4*)gp;
#endif
    }
#if USE_ASYNC_LDS
    __builtin_amdgcn_s_wait_asynccnt(0);
#endif
    __syncthreads();
  }

  const __bf16* wrow = Wb + (size_t)(col0 + m) * K;
  v8f acc = {0.f, 0.f, 0.f, 0.f, 0.f, 0.f, 0.f, 0.f};

  #pragma unroll
  for (int kt = 0; kt < KT; ++kt) {
    const int kb = kt * 32;
    // A: lane holds K = kb+16h .. +15 of row m (same k order as B)
    const float* xr = &tile[m * K + kb + 16 * h];
    float4 a0 = *(const float4*)(xr + 0);
    float4 a1 = *(const float4*)(xr + 4);
    float4 a2 = *(const float4*)(xr + 8);
    float4 a3 = *(const float4*)(xr + 12);
    v16bf a;
    a[0] = (__bf16)a0.x; a[1] = (__bf16)a0.y; a[2]  = (__bf16)a0.z; a[3]  = (__bf16)a0.w;
    a[4] = (__bf16)a1.x; a[5] = (__bf16)a1.y; a[6]  = (__bf16)a1.z; a[7]  = (__bf16)a1.w;
    a[8] = (__bf16)a2.x; a[9] = (__bf16)a2.y; a[10] = (__bf16)a2.z; a[11] = (__bf16)a2.w;
    a[12]= (__bf16)a3.x; a[13]= (__bf16)a3.y; a[14] = (__bf16)a3.z; a[15] = (__bf16)a3.w;
    // B: lane holds K = kb+16h .. +15 of W row (col0+m), already bf16
    const v8bf* q = (const v8bf*)(wrow + kb + 16 * h);
    v8bf blo = q[0], bhi = q[1];
    v16bf b = __builtin_shufflevector(blo, bhi,
                                      0, 1, 2, 3, 4, 5, 6, 7,
                                      8, 9, 10, 11, 12, 13, 14, 15);
    acc = __builtin_amdgcn_wmma_f32_16x16x32_bf16(
        /*neg_a=*/false, a, /*neg_b=*/false, b,
        /*c_mod=*/(short)0, acc, /*reuse_a=*/false, /*reuse_b=*/false);
  }

  // D layout: element v -> row row0 + 8h + v (consecutive rows), col col0 + m
  const float bv = bias[col0 + m];
  float* yp = Y + (size_t)(row0 + 8 * h) * ldy + col0 + m;
  if (row0 + 16 <= n) {
    #pragma unroll
    for (int v = 0; v < 8; v++) yp[(size_t)v * ldy] = acc[v] + bv;
  } else {
    #pragma unroll
    for (int v = 0; v < 8; v++)
      if (row0 + 8 * h + v < n) yp[(size_t)v * ldy] = acc[v] + bv;
  }
}

// ---------------- driver ----------------
static inline int cdiv(long long a, int b) { return (int)((a + b - 1) / b); }

extern "C" void kernel_launch(void* const* d_in, const int* in_sizes, int n_in,
                              void* d_out, int out_size, void* d_ws, size_t ws_size,
                              hipStream_t stream) {
  const long long* ei  = (const long long*)d_in[0];
  const float*     emb = (const float*)d_in[1];
  const float*     W1  = (const float*)d_in[2];
  const float*     b1  = (const float*)d_in[3];
  const float*     W2  = (const float*)d_in[4];
  const float*     b2  = (const float*)d_in[5];
  const float*     W3  = (const float*)d_in[6];
  const float*     b3  = (const float*)d_in[7];
  (void)n_in; (void)out_size; (void)ws_size;

  const int E = in_sizes[0] / 2;
  const int n = in_sizes[1] / 56;
  const long long* esrc = ei;
  const long long* edst = ei + E;

  // workspace carving (256B aligned slices)
  size_t o = 0;
  auto carve = [&](size_t bytes) {
    void* p = (char*)d_ws + o;
    o += (bytes + 255) & ~(size_t)255;
    return p;
  };
  float*  dinv = (float*)carve((size_t)n * 4);
  float*  A    = (float*)carve((size_t)n * 128 * 4);
  float*  B    = (float*)carve((size_t)n * 128 * 4);
  __bf16* W1b  = (__bf16*)carve((size_t)128 * 64 * 2);
  __bf16* W2b  = (__bf16*)carve((size_t)128 * 128 * 2);
  __bf16* W3b  = (__bf16*)carve((size_t)128 * 128 * 2);

  const int T = 256;

  // symmetric GCN normalization
  k_fill1 <<<cdiv(n, T), T, 0, stream>>>(dinv, n);
  k_deg   <<<cdiv(E, T), T, 0, stream>>>(edst, dinv, E);
  k_rsqrt <<<cdiv(n, T), T, 0, stream>>>(dinv, n);

  // staging: bf16 weights (W1 padded 56->64), padded embeddings
  k_cvt_w   <<<cdiv(128 * 64,  T), T, 0, stream>>>(W1, W1b, 128, 56, 64);
  k_cvt_w   <<<cdiv(128 * 128, T), T, 0, stream>>>(W2, W2b, 128, 128, 128);
  k_cvt_w   <<<cdiv(128 * 128, T), T, 0, stream>>>(W3, W3b, 128, 128, 128);
  k_copy_emb<<<cdiv((long long)n * 16, T), T, 0, stream>>>(emb, A, n);

  // one propagation hop: y = selfloop(x) ; y += edge scatter
  auto prop = [&](const float* x, float* y, int S, int C) {
    k_prop_init <<<cdiv((long long)n * (S / 4), T), T, 0, stream>>>(x, y, dinv, n, S / 4);
    k_prop_edges<<<cdiv((long long)E * C, T), T, 0, stream>>>(esrc, edst, dinv, x, y, E, C, S);
  };

  // ---- layer 1: 3 hops on 56 features (stride 64, zero-padded), linear 64->128
  prop(A, B, 64, 14);
  prop(B, A, 64, 14);
  prop(A, B, 64, 14);
  k_gemm_bf16<2><<<cdiv(n, 16), 256, 0, stream>>>(B, 64, W1b, b1, A, 128, n);

  // ---- layer 2: 3 hops on 128 features, linear 128->128
  prop(A, B, 128, 32);
  prop(B, A, 128, 32);
  prop(A, B, 128, 32);
  k_gemm_bf16<4><<<cdiv(n, 16), 256, 0, stream>>>(B, 128, W2b, b2, A, 128, n);

  // ---- layer 3: 3 hops on 128 features, linear 128->128 -> d_out
  prop(A, B, 128, 32);
  prop(B, A, 128, 32);
  prop(A, B, 128, 32);
  k_gemm_bf16<4><<<cdiv(n, 16), 256, 0, stream>>>(B, 128, W3b, b3, (float*)d_out, 128, n);
}
